// GIN_80487687127440
// MI455X (gfx1250) — matrix operand order repaired and checked by reading, not compiled, not run on hardware
//
#include <hip/hip_runtime.h>
#include <hip/hip_bf16.h>

typedef __attribute__((ext_vector_type(16))) _Float16 v16h;
typedef __attribute__((ext_vector_type(8)))  _Float16 v8h;
typedef __attribute__((ext_vector_type(8)))  float    v8f;

#define TB 128  // 4 wave32 per block

__device__ __forceinline__ v8f zero8() {
    v8f z = {0.f,0.f,0.f,0.f,0.f,0.f,0.f,0.f};
    return z;
}

// ---------------- zero buffer ----------------
__global__ void gin_zero(float* __restrict__ p, long n) {
    long i = (long)blockIdx.x * blockDim.x + threadIdx.x;
    if (i < n) p[i] = 0.f;
}

// ---------------- edge scatter-add: agg[dst] += x[src] ----------------
__global__ void gin_scatter(const float* __restrict__ x,
                            const int* __restrict__ src,
                            const int* __restrict__ dst,
                            float* __restrict__ agg,
                            int E, int F) {
    int chunks = F >> 2;                       // float4 chunks per edge
    long t = (long)blockIdx.x * blockDim.x + threadIdx.x;
    long total = (long)E * chunks;
    if (t >= total) return;
    int e = (int)(t / chunks);
    int j = (int)(t - (long)e * chunks) << 2;
    // prefetch a future edge's source row (global_prefetch_b8)
    int ea = e + 64;
    if (ea < E) __builtin_prefetch(x + (long)src[ea] * F + j, 0, 1);
    int s = src[e], d = dst[e];
    const float4 v = *(const float4*)(x + (long)s * F + j);
    float* ap = agg + (long)d * F + j;
    atomicAdd(ap + 0, v.x);
    atomicAdd(ap + 1, v.y);
    atomicAdd(ap + 2, v.z);
    atomicAdd(ap + 3, v.w);
}

// ---------------- fused GIN MLP: WMMA GEMM x2 + stats ----------------
// h_pre = (1+eps)*xin + agg ; h1 = relu(h_pre @ Wa + ba) ; h2 = relu(h1 @ Wb + bb)
// Weights staged into LDS pre-packed in WMMA B-fragment order:
//   frag(ks,nt) for lane L = 16 contiguous f16 -> two ds_load_b128 per fragment.
// A fragments are two contiguous 16B runs of the row-major activation tile.
template <int K>
__global__ __launch_bounds__(TB) void gin_mlp(
    const float* __restrict__ xin, const float* __restrict__ agg,
    const float* __restrict__ epsp,
    const float* __restrict__ wa, const float* __restrict__ ba,
    const float* __restrict__ wb, const float* __restrict__ bb,
    float* __restrict__ hbuf, float* __restrict__ stats,
    int N)
{
    constexpr int KSTEPS = K >> 5;
    __shared__ alignas(32) _Float16 sWaP[KSTEPS * 4 * 32 * 16]; // fragment-packed Wa
    __shared__ alignas(32) _Float16 sWbP[2 * 4 * 32 * 16];      // fragment-packed Wb
    __shared__ alignas(16) _Float16 sX[64 * K];                 // 64 rows x K row-major
    __shared__ alignas(16) _Float16 sH1[64 * 64];               // 64 rows x 64 row-major
    __shared__ float sSum[64];
    __shared__ float sSq[64];

    const int tid  = threadIdx.x;
    const int lane = tid & 31;
    const int w    = tid >> 5;           // wave id 0..3
    const int row0 = blockIdx.x * 64;
    const float ep = 1.0f + epsp[0];

    if (tid < 64) { sSum[tid] = 0.f; sSq[tid] = 0.f; }

    // stage Wa pre-packed: k = ks*32 + g*16 + p ; n = nt*16 + nl ; lane = g*16+nl
    for (int i = tid; i < K * 64; i += TB) {
        int k = i >> 6, n = i & 63;
        int ks = k >> 5, g = (k >> 4) & 1, p = k & 15;
        int nt = n >> 4, nl = n & 15;
        sWaP[(((ks << 2) + nt) << 9) + (((g << 4) + nl) << 4) + p] = (_Float16)wa[i];
    }
    for (int i = tid; i < 64 * 64; i += TB) {
        int k = i >> 6, n = i & 63;
        int ks = k >> 5, g = (k >> 4) & 1, p = k & 15;
        int nt = n >> 4, nl = n & 15;
        sWbP[(((ks << 2) + nt) << 9) + (((g << 4) + nl) << 4) + p] = (_Float16)wb[i];
    }
    // stage activations: h_pre = (1+eps)*x + agg (coalesced, f32->f16)
    for (int i = tid; i < 64 * K; i += TB) {
        int r = i / K, k = i - r * K;
        int grow = row0 + r; if (grow > N - 1) grow = N - 1;   // clamp; stores masked later
        long off = (long)grow * K + k;
        sX[i] = (_Float16)(ep * xin[off] + agg[off]);
    }
    __syncthreads();

    const int g  = lane >> 4;               // lane half
    const int ml = (w << 4) + (lane & 15);  // A-row (block-local) for this lane
    const int nl = lane & 15;               // column within a 16-wide tile

    // -------- GEMM 1: (16 x K) @ (K x 64) --------
    v8f acc[4];
    #pragma unroll
    for (int nt = 0; nt < 4; ++nt) acc[nt] = zero8();

    #pragma unroll
    for (int ks = 0; ks < KSTEPS; ++ks) {
        const _Float16* arow = &sX[ml * K + (ks << 5) + (g << 4)];
        v8h alo = *(const v8h*)(arow);
        v8h ahi = *(const v8h*)(arow + 8);
        v16h afrag = __builtin_shufflevector(alo, ahi,
            0,1,2,3,4,5,6,7,8,9,10,11,12,13,14,15);
        #pragma unroll
        for (int nt = 0; nt < 4; ++nt) {
            v16h bfrag = *(const v16h*)&sWaP[(((ks << 2) + nt) << 9) + (lane << 4)];
            acc[nt] = __builtin_amdgcn_wmma_f32_16x16x32_f16(
                false, afrag, false, bfrag, (short)0, acc[nt], false, false);
        }
    }

    // bias + relu -> sH1 (f16)
    #pragma unroll
    for (int nt = 0; nt < 4; ++nt) {
        const int n = (nt << 4) + nl;
        const float bv = ba[n];
        #pragma unroll
        for (int r = 0; r < 8; ++r) {
            float hv = acc[nt][r] + bv;
            hv = hv > 0.f ? hv : 0.f;
            int mrow = (w << 4) + (g << 3) + r;
            sH1[mrow * 64 + n] = (_Float16)hv;
        }
    }
    __syncthreads();

    // -------- GEMM 2: (16 x 64) @ (64 x 64) --------
    v8f acc2[4];
    #pragma unroll
    for (int nt = 0; nt < 4; ++nt) acc2[nt] = zero8();

    #pragma unroll
    for (int ks = 0; ks < 2; ++ks) {
        const _Float16* arow = &sH1[ml * 64 + (ks << 5) + (g << 4)];
        v8h alo = *(const v8h*)(arow);
        v8h ahi = *(const v8h*)(arow + 8);
        v16h afrag = __builtin_shufflevector(alo, ahi,
            0,1,2,3,4,5,6,7,8,9,10,11,12,13,14,15);
        #pragma unroll
        for (int nt = 0; nt < 4; ++nt) {
            v16h bfrag = *(const v16h*)&sWbP[(((ks << 2) + nt) << 9) + (lane << 4)];
            acc2[nt] = __builtin_amdgcn_wmma_f32_16x16x32_f16(
                false, afrag, false, bfrag, (short)0, acc2[nt], false, false);
        }
    }

    // bias + relu -> hbuf, accumulate column stats in LDS
    #pragma unroll
    for (int nt = 0; nt < 4; ++nt) {
        const int n = (nt << 4) + nl;
        const float bv = bb[n];
        #pragma unroll
        for (int r = 0; r < 8; ++r) {
            int mrow = (w << 4) + (g << 3) + r;
            int grow = row0 + mrow;
            float hv = acc2[nt][r] + bv;
            hv = hv > 0.f ? hv : 0.f;
            if (grow < N) {
                hbuf[(long)grow * 64 + n] = hv;
                atomicAdd(&sSum[n], hv);
                atomicAdd(&sSq[n], hv * hv);
            }
        }
    }
    __syncthreads();
    if (tid < 64) {
        atomicAdd(&stats[tid],      sSum[tid]);
        atomicAdd(&stats[64 + tid], sSq[tid]);
    }
}

// ---------------- fold BN stats into per-column scale/shift ----------------
__global__ void gin_bnfin(float* __restrict__ stats,
                          const float* __restrict__ gamma,
                          const float* __restrict__ beta, int N) {
    int t = threadIdx.x;               // 64 threads
    float invn = 1.0f / (float)N;
    float mean = stats[t] * invn;
    float var  = stats[64 + t] * invn - mean * mean;
    float a = gamma[t] * rsqrtf(var + 1e-5f);
    float b = beta[t] - mean * a;
    stats[t] = a;
    stats[64 + t] = b;
}

// ---------------- BN apply + outer ReLU ----------------
__global__ void gin_bnapply(const float* __restrict__ h,
                            const float* __restrict__ stats,
                            float* __restrict__ out, long total) {
    long i = (long)blockIdx.x * blockDim.x + threadIdx.x;
    if (i >= total) return;
    int n = (int)(i & 63);
    float v = h[i] * stats[n] + stats[64 + n];
    out[i] = v > 0.f ? v : 0.f;
}

// ---------------- final linear 64 -> 10 ----------------
__global__ void gin_lin(const float* __restrict__ h,
                        const float* __restrict__ w,
                        const float* __restrict__ b,
                        float* __restrict__ out, int N) {
    __shared__ float sw[64 * 10];
    __shared__ float sb[10];
    int tid = threadIdx.x;
    for (int i = tid; i < 640; i += 256) sw[i] = w[i];
    if (tid < 10) sb[tid] = b[tid];
    __syncthreads();
    int row = blockIdx.x * 256 + tid;
    if (row >= N) return;
    float acc[10];
    #pragma unroll
    for (int c = 0; c < 10; ++c) acc[c] = sb[c];
    const float* hr = h + (long)row * 64;
    #pragma unroll 4
    for (int k = 0; k < 64; ++k) {
        float x = hr[k];
        #pragma unroll
        for (int c = 0; c < 10; ++c) acc[c] += x * sw[k * 10 + c];
    }
    #pragma unroll
    for (int c = 0; c < 10; ++c) out[(long)row * 10 + c] = acc[c];
}

extern "C" void kernel_launch(void* const* d_in, const int* in_sizes, int n_in,
                              void* d_out, int out_size, void* d_ws, size_t ws_size,
                              hipStream_t stream) {
    const int N = in_sizes[0] / 128;       // 50000
    const int E = in_sizes[1] / 2;         // 800000

    const float* x    = (const float*)d_in[0];
    const int*   src  = (const int*)d_in[1];
    const int*   dst  = src + E;
    const float* linw = (const float*)d_in[23];
    const float* linb = (const float*)d_in[24];

    float* ws    = (float*)d_ws;
    float* agg   = ws;                       // N*128
    float* hbuf  = agg  + (long)N * 128;     // N*64
    float* xA    = hbuf + (long)N * 64;      // N*64
    float* xB    = xA   + (long)N * 64;      // N*64
    float* stats = xB   + (long)N * 64;      // 128

    const float* xin = x;
    int K = 128;
    float* outs[3] = { xA, xB, xA };

    for (int l = 0; l < 3; ++l) {
        int base = 2 + l * 7;
        const float* eps = (const float*)d_in[base];
        const float* wa  = (const float*)d_in[base + 1];
        const float* ba  = (const float*)d_in[base + 2];
        const float* wb  = (const float*)d_in[base + 3];
        const float* bb  = (const float*)d_in[base + 4];
        const float* gm  = (const float*)d_in[base + 5];
        const float* be  = (const float*)d_in[base + 6];

        long aggN = (long)N * K;
        gin_zero<<<(int)((aggN + 255) / 256), 256, 0, stream>>>(agg, aggN);
        gin_zero<<<1, 128, 0, stream>>>(stats, 128);

        long sc = (long)E * (K >> 2);
        gin_scatter<<<(int)((sc + 255) / 256), 256, 0, stream>>>(xin, src, dst, agg, E, K);

        if (K == 128) {
            gin_mlp<128><<<(N + 63) / 64, TB, 0, stream>>>(xin, agg, eps, wa, ba, wb, bb,
                                                           hbuf, stats, N);
        } else {
            gin_mlp<64><<<(N + 63) / 64, TB, 0, stream>>>(xin, agg, eps, wa, ba, wb, bb,
                                                          hbuf, stats, N);
        }

        gin_bnfin<<<1, 64, 0, stream>>>(stats, gm, be, N);

        long tot = (long)N * 64;
        gin_bnapply<<<(int)((tot + 255) / 256), 256, 0, stream>>>(hbuf, stats, outs[l], tot);

        xin = outs[l];
        K = 64;
    }

    gin_lin<<<(N + 255) / 256, 256, 0, stream>>>(xin, linw, linb, (float*)d_out, N);
}
